// HebbianBlock_88098369175975
// MI455X (gfx1250) — compile-verified, matrix operand order checked by reading, not run on hardware
//
#include <hip/hip_runtime.h>
#include <hip/hip_bf16.h>
#include <math.h>

// ---------------------------------------------------------------------------
// Types for CDNA5 (gfx1250) WMMA
// ---------------------------------------------------------------------------
typedef __bf16 bf16;
typedef __attribute__((ext_vector_type(16))) __bf16 v16bf;
typedef __attribute__((ext_vector_type(8)))  __bf16 v8bf;
typedef __attribute__((ext_vector_type(8)))  float  v8f;

union FragU {
    v16bf v;
    v8bf  h[2];
};

__device__ __forceinline__ v8f wmma_bf16(v16bf a, v16bf b, v8f c) {
    // D = A(16x32 bf16) x B(32x16 bf16) + C(16x16 f32)
    return __builtin_amdgcn_wmma_f32_16x16x32_bf16(
        /*neg_a=*/false, a, /*neg_b=*/false, b,
        /*c_mod=*/(short)0, c, /*reuse_a=*/false, /*reuse_b=*/false);
}

__device__ __forceinline__ float silu(float v) {
    return v / (1.0f + expf(-v));
}

// ---------------------------------------------------------------------------
// gfx1250 async global->LDS copy path (ASYNCcnt), guarded so the file still
// compiles (sync fallback) if this toolchain lacks the builtins.
// The b128 builtin takes GCC-vector int4 pointers: (v4i AS1*, v4i AS3*, Ii, Ii).
// ---------------------------------------------------------------------------
#if defined(__AMDGCN__) &&                                                    \
    __has_builtin(__builtin_amdgcn_global_load_async_to_lds_b128) &&          \
    __has_builtin(__builtin_amdgcn_s_wait_asynccnt)
#define HAVE_ASYNC_LDS 1
#else
#define HAVE_ASYNC_LDS 0
#endif

#define AS1 __attribute__((address_space(1)))
#define AS3 __attribute__((address_space(3)))

typedef int v4i_gcc __attribute__((vector_size(4 * sizeof(int))));

__device__ __forceinline__ void async_b128(const bf16* g, bf16* l) {
#if HAVE_ASYNC_LDS
    __builtin_amdgcn_global_load_async_to_lds_b128(
        (AS1 v4i_gcc*)g, (AS3 v4i_gcc*)l, /*offset=*/0, /*cpol=*/0);
#else
    *reinterpret_cast<v8bf*>(l) = *reinterpret_cast<const v8bf*>(g);
#endif
}

template <int N>
__device__ __forceinline__ void async_wait() {
#if HAVE_ASYNC_LDS
    __builtin_amdgcn_s_wait_asynccnt(N);
#endif
}

// Problem constants (match reference)
#define BB   16
#define TT   1024
#define DD   768
#define DI   3072
#define MM   (BB * TT)   // 16384

// ---------------------------------------------------------------------------
// fp32 -> bf16 elementwise cast (weights)
// ---------------------------------------------------------------------------
__global__ void cvt_f32_bf16_kernel(const float* __restrict__ in,
                                    bf16* __restrict__ out, int n) {
    int i = blockIdx.x * blockDim.x + threadIdx.x;
    if (i < n) out[i] = (bf16)in[i];
}

// ---------------------------------------------------------------------------
// RMSNorm + bf16 cast.  One block per row of [B*T, D].
// ---------------------------------------------------------------------------
__global__ __launch_bounds__(256)
void rmsnorm_kernel(const float* __restrict__ x, const float* __restrict__ w,
                    bf16* __restrict__ h, int Dn) {
    __shared__ float sbuf[256];
    const int row = blockIdx.x;
    const int tid = threadIdx.x;
    const float* xr = x + (size_t)row * Dn;

    float ss = 0.f;
    for (int i = tid; i < Dn; i += 256) { float v = xr[i]; ss += v * v; }
    sbuf[tid] = ss;
    __syncthreads();
    for (int s = 128; s > 0; s >>= 1) {
        if (tid < s) sbuf[tid] += sbuf[tid + s];
        __syncthreads();
    }
    const float rinv = rsqrtf(sbuf[0] / (float)Dn + 1e-5f);
    bf16* hr = h + (size_t)row * Dn;
    for (int i = tid; i < Dn; i += 256) hr[i] = (bf16)(xr[i] * rinv * w[i]);
}

// ---------------------------------------------------------------------------
// Generic bf16 WMMA GEMM:  C[M,N] = A[M,K] * W[N,K]^T   (weights row-major N,K)
// Block tile 64(M) x 256(N), K-step 32, double-buffered LDS with async
// global->LDS copies (GLOBAL_LOAD_ASYNC_TO_LDS_B128 / s_wait_asynccnt).
// 8 wave32 waves: 2(M) x 4(N); each wave 32x64 via 2x4 WMMA fragments.
// EPI==0 : store bf16 C (row index z*M+row)
// EPI==1 : d_out = x + out0 + exp(log_alpha)*acc   (fp32 store)
// ---------------------------------------------------------------------------
#define LDA 48   // padded LDS row stride in bf16 (96B, 16B aligned)

template <int EPI>
__global__ __launch_bounds__(256)
void gemm_bf16_kernel(const bf16* __restrict__ A, long long strideAbatch,
                      const bf16* __restrict__ W, long long strideWbatch,
                      bf16* __restrict__ Cbf,
                      const float* __restrict__ Xres,
                      const bf16* __restrict__ Out0,
                      const float* __restrict__ logAlpha,
                      float* __restrict__ Cf,
                      int M, int N, int K) {
    __shared__ bf16 As[2][64 * LDA];
    __shared__ bf16 Bs[2][256 * LDA];

    const int tid  = threadIdx.x;
    const int lane = tid & 31;
    const int wid  = tid >> 5;        // 0..7
    const int wm   = wid & 1;         // 0..1  (M)
    const int wn   = wid >> 1;        // 0..3  (N)
    const int half = lane >> 4;       // 0/1
    const int lrow = lane & 15;

    const int z  = blockIdx.z;
    const int n0 = blockIdx.x * 256;
    const int m0 = blockIdx.y * 64;

    const bf16* Ab = A + (long long)z * strideAbatch;
    const bf16* Wb = W + (long long)z * strideWbatch;

    const v8f vzero = {0.f, 0.f, 0.f, 0.f, 0.f, 0.f, 0.f, 0.f};
    v8f acc[2][4];
#pragma unroll
    for (int i = 0; i < 2; ++i)
#pragma unroll
        for (int j = 0; j < 4; ++j) acc[i][j] = vzero;

    const int ar = tid >> 2;          // A tile row     0..63
    const int ac = tid & 3;           // A tile chunk   0..3  (8 bf16 each)

    // Stage one 64x32 A tile + 256x32 B tile into LDS buffer `buf`.
    // Each thread issues exactly 5 async b128 copies (1 A + 4 B).
    auto stage = [&](int buf, int k0) {
        async_b128(Ab + (size_t)(m0 + ar) * K + k0 + ac * 8,
                   &As[buf][ar * LDA + ac * 8]);
        const bf16* src = Wb + (size_t)(n0 + tid) * K + k0;
        bf16* dst = &Bs[buf][tid * LDA];
#pragma unroll
        for (int c = 0; c < 4; ++c)
            async_b128(src + c * 8, dst + c * 8);
    };

    const int nk = K / 32;
    stage(0, 0);

    for (int kb = 0; kb < nk; ++kb) {
        const int buf = kb & 1;
        __syncthreads();              // everyone done reading buf^1
        if (kb + 1 < nk) {
            stage(buf ^ 1, (kb + 1) * 32);   // in flight during compute of buf
            async_wait<5>();                  // retire current tile's 5 copies
        } else {
            async_wait<0>();
        }
        __syncthreads();              // current tile visible to all waves

        // ---- fragments from LDS (ISA 16-bit A/B layouts) ----
        FragU a[2], b[4];
#pragma unroll
        for (int i = 0; i < 2; ++i) {
            const bf16* p = &As[buf][(wm * 32 + i * 16 + lrow) * LDA + half * 8];
            a[i].h[0] = *reinterpret_cast<const v8bf*>(p);
            a[i].h[1] = *reinterpret_cast<const v8bf*>(p + 16);
        }
#pragma unroll
        for (int j = 0; j < 4; ++j) {
            const bf16* p = &Bs[buf][(wn * 64 + j * 16 + lrow) * LDA + half * 8];
            b[j].h[0] = *reinterpret_cast<const v8bf*>(p);
            b[j].h[1] = *reinterpret_cast<const v8bf*>(p + 16);
        }
#pragma unroll
        for (int i = 0; i < 2; ++i)
#pragma unroll
            for (int j = 0; j < 4; ++j)
                acc[i][j] = wmma_bf16(a[i].v, b[j].v, acc[i][j]);
    }

    // ---- epilogue ----
    float alpha = 0.f;
    if (EPI == 1) alpha = expf(logAlpha[0]);

#pragma unroll
    for (int i = 0; i < 2; ++i) {
#pragma unroll
        for (int j = 0; j < 4; ++j) {
            const int col = n0 + wn * 64 + j * 16 + lrow;
#pragma unroll
            for (int r = 0; r < 8; ++r) {
                const int row = m0 + wm * 32 + i * 16 + half * 8 + r;
                const size_t idx = ((size_t)z * M + row) * (size_t)N + col;
                if (EPI == 0) {
                    Cbf[idx] = (bf16)acc[i][j][r];
                } else {
                    Cf[idx] = Xres[idx] + (float)Out0[idx] + alpha * acc[i][j][r];
                }
            }
        }
    }
}

// ---------------------------------------------------------------------------
// Depthwise 3x3 conv (pad 1) + bias + SiLU, then multiply by SiLU(gate).
// One block per (batch, channel); 32x32 spatial tile staged in LDS.
// In-place: val -> gated.
// ---------------------------------------------------------------------------
__global__ __launch_bounds__(256)
void dwconv_gate_kernel(bf16* __restrict__ val, const bf16* __restrict__ gate,
                        const float* __restrict__ cw, const float* __restrict__ cb,
                        int dI) {
    __shared__ float tile[34 * 34];
    const int be = blockIdx.x;
    const int e  = be % dI;
    const int b  = be / dI;
    const size_t base = (size_t)b * TT * dI + e;
    const int tid = threadIdx.x;

    for (int i = tid; i < 34 * 34; i += 256) tile[i] = 0.f;
    __syncthreads();
    for (int p = tid; p < 1024; p += 256) {
        const int hh = p >> 5, ww = p & 31;
        tile[(hh + 1) * 34 + (ww + 1)] = (float)val[base + (size_t)p * dI];
    }
    __syncthreads();

    float w9[9];
#pragma unroll
    for (int q = 0; q < 9; ++q) w9[q] = cw[e * 9 + q];
    const float bias = cb[e];

    for (int p = tid; p < 1024; p += 256) {
        const int hh = p >> 5, ww = p & 31;
        float a = bias;
#pragma unroll
        for (int kh = 0; kh < 3; ++kh)
#pragma unroll
            for (int kw = 0; kw < 3; ++kw)
                a += tile[(hh + kh) * 34 + (ww + kw)] * w9[kh * 3 + kw];
        const float g = (float)gate[base + (size_t)p * dI];
        val[base + (size_t)p * dI] = (bf16)(silu(a) * silu(g));
    }
}

// ---------------------------------------------------------------------------
// Batched covariance: Wout[b,d,e] = (1/T) * sum_t K[b,t,d] * V[b,t,e]
// Block tile 64(d) x 64(e), K-step 32(t).  K/V tiles transposed through LDS.
// 8 waves: 2(d) x 4(e); each wave 32x16 via 2 WMMA fragments.
// ---------------------------------------------------------------------------
__global__ __launch_bounds__(256)
void cov_kernel(const bf16* __restrict__ Km, const bf16* __restrict__ Vm,
                bf16* __restrict__ Wout, int Dn, int T) {
    __shared__ bf16 KT[64 * LDA];
    __shared__ bf16 VT[64 * LDA];

    const int tid  = threadIdx.x;
    const int lane = tid & 31;
    const int wid  = tid >> 5;
    const int wm   = wid & 1;
    const int wn   = wid >> 1;
    const int half = lane >> 4;
    const int lrow = lane & 15;

    const int e0 = blockIdx.x * 64;
    const int d0 = blockIdx.y * 64;
    const int b  = blockIdx.z;

    const bf16* Kb = Km + (size_t)b * T * Dn;
    const bf16* Vb = Vm + (size_t)b * T * Dn;

    const v8f vzero = {0.f, 0.f, 0.f, 0.f, 0.f, 0.f, 0.f, 0.f};
    v8f acc[2] = {vzero, vzero};

    const int tt = tid >> 3;   // 0..31 (t within tile)
    const int cc = tid & 7;    // 0..7  (8-element d/e chunk)

    for (int t0 = 0; t0 < T; t0 += 32) {
        const v8bf kv = *reinterpret_cast<const v8bf*>(
            Kb + (size_t)(t0 + tt) * Dn + d0 + cc * 8);
        const v8bf vv = *reinterpret_cast<const v8bf*>(
            Vb + (size_t)(t0 + tt) * Dn + e0 + cc * 8);
#pragma unroll
        for (int q = 0; q < 8; ++q) {
            KT[(cc * 8 + q) * LDA + tt] = kv[q];
            VT[(cc * 8 + q) * LDA + tt] = vv[q];
        }
        __syncthreads();

        FragU a[2], bb;
#pragma unroll
        for (int i = 0; i < 2; ++i) {
            const bf16* p = &KT[(wm * 32 + i * 16 + lrow) * LDA + half * 8];
            a[i].h[0] = *reinterpret_cast<const v8bf*>(p);
            a[i].h[1] = *reinterpret_cast<const v8bf*>(p + 16);
        }
        {
            const bf16* p = &VT[(wn * 16 + lrow) * LDA + half * 8];
            bb.h[0] = *reinterpret_cast<const v8bf*>(p);
            bb.h[1] = *reinterpret_cast<const v8bf*>(p + 16);
        }
#pragma unroll
        for (int i = 0; i < 2; ++i) acc[i] = wmma_bf16(a[i].v, bb.v, acc[i]);
        __syncthreads();
    }

    const float invT = 1.0f / (float)T;
#pragma unroll
    for (int i = 0; i < 2; ++i) {
        const int ecol = e0 + wn * 16 + lrow;
#pragma unroll
        for (int r = 0; r < 8; ++r) {
            const int drow = d0 + wm * 32 + i * 16 + half * 8 + r;
            Wout[(size_t)b * Dn * Dn + (size_t)drow * Dn + ecol] =
                (bf16)(acc[i][r] * invT);
        }
    }
}

// ---------------------------------------------------------------------------
// Host-side launch sequence
// ---------------------------------------------------------------------------
extern "C" void kernel_launch(void* const* d_in, const int* in_sizes, int n_in,
                              void* d_out, int out_size, void* d_ws, size_t ws_size,
                              hipStream_t stream) {
    (void)in_sizes; (void)n_in; (void)out_size; (void)ws_size;

    const float* x        = (const float*)d_in[0];
    const float* norm_w   = (const float*)d_in[1];
    const float* w_proj   = (const float*)d_in[2];
    const float* w_gate   = (const float*)d_in[3];
    const float* conv_w   = (const float*)d_in[4];
    const float* conv_b   = (const float*)d_in[5];
    const float* w_out    = (const float*)d_in[6];
    const float* w_k      = (const float*)d_in[7];
    const float* w_v      = (const float*)d_in[8];
    const float* log_alpha= (const float*)d_in[9];
    float* out = (float*)d_out;

    // workspace carve-out (256B aligned)
    char* wsp = (char*)d_ws;
    auto alloc = [&](size_t bytes) -> char* {
        char* p = wsp;
        wsp += (bytes + 255) & ~(size_t)255;
        return p;
    };
    bf16* h_bf  = (bf16*)alloc((size_t)MM * DD * 2);
    bf16* val   = (bf16*)alloc((size_t)MM * DI * 2);
    bf16* gatel = (bf16*)alloc((size_t)MM * DI * 2);
    bf16* out0  = (bf16*)alloc((size_t)MM * DD * 2);
    bf16* Kmat  = (bf16*)alloc((size_t)MM * DD * 2);
    bf16* Vmat  = (bf16*)alloc((size_t)MM * DD * 2);
    bf16* Wcov  = (bf16*)alloc((size_t)BB * DD * DD * 2);
    bf16* wproj_b = (bf16*)alloc((size_t)DI * DD * 2);
    bf16* wgate_b = (bf16*)alloc((size_t)DI * DD * 2);
    bf16* wout_b  = (bf16*)alloc((size_t)DD * DI * 2);
    bf16* wk_b    = (bf16*)alloc((size_t)DD * DD * 2);
    bf16* wv_b    = (bf16*)alloc((size_t)DD * DD * 2);

    // 1) weight casts fp32 -> bf16
    auto cvt = [&](const float* src, bf16* dst, int n) {
        cvt_f32_bf16_kernel<<<(n + 255) / 256, 256, 0, stream>>>(src, dst, n);
    };
    cvt(w_proj, wproj_b, DI * DD);
    cvt(w_gate, wgate_b, DI * DD);
    cvt(w_out,  wout_b,  DD * DI);
    cvt(w_k,    wk_b,    DD * DD);
    cvt(w_v,    wv_b,    DD * DD);

    // 2) RMSNorm -> h (bf16)
    rmsnorm_kernel<<<MM, 256, 0, stream>>>(x, norm_w, h_bf, DD);

    // 3) val = h @ w_proj^T   [16384 x 3072]
    gemm_bf16_kernel<0><<<dim3(DI / 256, MM / 64, 1), 256, 0, stream>>>(
        h_bf, 0, wproj_b, 0, val, nullptr, nullptr, nullptr, nullptr,
        MM, DI, DD);

    // 4) gate = h @ w_gate^T  [16384 x 3072]
    gemm_bf16_kernel<0><<<dim3(DI / 256, MM / 64, 1), 256, 0, stream>>>(
        h_bf, 0, wgate_b, 0, gatel, nullptr, nullptr, nullptr, nullptr,
        MM, DI, DD);

    // 5) depthwise conv + SiLU + gate (in-place: val -> gated)
    dwconv_gate_kernel<<<BB * DI, 256, 0, stream>>>(val, gatel, conv_w, conv_b, DI);

    // 6) out0 = gated @ w_out^T  [16384 x 768]
    gemm_bf16_kernel<0><<<dim3(DD / 256, MM / 64, 1), 256, 0, stream>>>(
        val, 0, wout_b, 0, out0, nullptr, nullptr, nullptr, nullptr,
        MM, DD, DI);

    // 7) K = h @ w_k^T   [16384 x 768]
    gemm_bf16_kernel<0><<<dim3(DD / 256, MM / 64, 1), 256, 0, stream>>>(
        h_bf, 0, wk_b, 0, Kmat, nullptr, nullptr, nullptr, nullptr,
        MM, DD, DD);

    // 8) V = out0 @ w_v^T  [16384 x 768]
    gemm_bf16_kernel<0><<<dim3(DD / 256, MM / 64, 1), 256, 0, stream>>>(
        out0, 0, wv_b, 0, Vmat, nullptr, nullptr, nullptr, nullptr,
        MM, DD, DD);

    // 9) Wcov[b] = K[b]^T V[b] / T   [16 x 768 x 768]
    cov_kernel<<<dim3(DD / 64, DD / 64, BB), 256, 0, stream>>>(
        Kmat, Vmat, Wcov, DD, TT);

    // 10) d_out = x + out0 + exp(log_alpha) * (K[b] @ Wcov[b]^T)
    gemm_bf16_kernel<1><<<dim3(DD / 256, TT / 64, BB), 256, 0, stream>>>(
        Kmat, (long long)TT * DD, Wcov, (long long)DD * DD,
        nullptr, x, out0, log_alpha, out,
        TT, DD, DD);
}
